// SparseCoder_82119774699827
// MI455X (gfx1250) — matrix operand order repaired and checked by reading, not compiled, not run on hardware
//
#include <hip/hip_runtime.h>
#include <hip/hip_bf16.h>

// ---------------------------------------------------------------------------
// TopK Sparse Autoencoder forward for MI455X (gfx1250, wave32, WMMA + TDM).
//
// Encoder GEMM (4096x1024 @ 1024x65536, ~550 GFLOP) runs as split-bf16:
// a = a_hi + a_lo (bf16 pair), C += 3 WMMAs (hi*hi + hi*lo + lo*hi) per
// 16x16x32 tile -> ~2^-17 relative error (fp32-class, safe for exact top-k).
//
// All f32->bf16 conversion is hoisted OUT of the GEMM inner loop:
//  * x_c = x - b_dec is pre-split into packed bf16 hi/lo arrays in ws (16MB,
//    L2-resident); A tiles are TDM-loaded (tensor_load_to_lds, data_size=2B,
//    padded rows) and A fragments are two ds_load_b128 each.
//  * W_enc tiles are cooperatively split+transposed into bf16 LDS once per
//    block-stage (256 threads, coalesced global reads prefetched one stage
//    ahead); B fragments are two ds_load_b128 each.
// Inner loop per wave-stage: ~16 ds_load_b128 + 12 v_wmma + ~25 VALU.
// Grid is M-fastest so co-resident blocks share W_enc columns: W_enc (256MB)
// streams from HBM once. pre_acts (1GB) uses non-temporal stores/loads.
// ---------------------------------------------------------------------------

#define D_IN   1024
#define NLAT   65536
#define BATCH  4096
#define TOPK   32

typedef __attribute__((ext_vector_type(16))) __bf16          v16bf;
typedef __attribute__((ext_vector_type(16))) unsigned short  v16us;
typedef __attribute__((ext_vector_type(8)))  unsigned short  v8us;
typedef __attribute__((ext_vector_type(2)))  __bf16          v2bf;
typedef __attribute__((ext_vector_type(8)))  float           v8f;
typedef __attribute__((ext_vector_type(4)))  unsigned int    v4u;
typedef __attribute__((ext_vector_type(8)))  int             v8i;
typedef __attribute__((ext_vector_type(4)))  int             v4i;

union V16 { v16us u; v16bf b; v8us h[2]; };
union U2  { v2bf v; unsigned u; };

#if defined(__has_builtin)
#  if __has_builtin(__builtin_amdgcn_cvt_pk_bf16_f32)
#    define HAVE_CVT_PK_BF16 1
#  endif
#endif
#ifndef HAVE_CVT_PK_BF16
#  define HAVE_CVT_PK_BF16 0
#endif

__device__ __forceinline__ unsigned short f32_to_bf16_rn(float f) {
  unsigned int u = __float_as_uint(f);
  u += 0x7fffu + ((u >> 16) & 1u);      // round-to-nearest-even
  return (unsigned short)(u >> 16);
}

// pack two floats to two bf16 (RNE), low = a, high = b
__device__ __forceinline__ unsigned pack_bf16_rn(float a, float b) {
#if HAVE_CVT_PK_BF16
  U2 t; t.v = __builtin_amdgcn_cvt_pk_bf16_f32(a, b);
  return t.u;
#else
  return (unsigned)f32_to_bf16_rn(a) | ((unsigned)f32_to_bf16_rn(b) << 16);
#endif
}

// split (a,b) into packed bf16 hi dword and residual-lo dword
__device__ __forceinline__ void split_pair(float a, float b,
                                           unsigned& hi, unsigned& lo) {
  hi = pack_bf16_rn(a, b);
  float ha = __uint_as_float(hi << 16);
  float hb = __uint_as_float(hi & 0xffff0000u);
  lo = pack_bf16_rn(a - ha, b - hb);
}

// ---------------------------------------------------------------------------
// TDM: 2-D tile load (global -> LDS) with LDS row padding (6-arg builtin).
// ---------------------------------------------------------------------------
__device__ __forceinline__ void tdm_load_2d(
    unsigned lds_off, const void* gptr, unsigned data_size_code,
    unsigned tensor_d0, unsigned tensor_d1,
    unsigned tile_d0, unsigned tile_d1,
    unsigned long long stride0_elems,
    unsigned pad_interval_code, unsigned pad_amount_code)
{
  unsigned long long ga = (unsigned long long)gptr;
  v4u g0;
  g0[0] = 1u;                                             // count=1 (valid)
  g0[1] = lds_off;                                        // LDS byte addr
  g0[2] = (unsigned)(ga & 0xffffffffull);                 // global addr lo
  g0[3] = (unsigned)((ga >> 32) & 0x01ffffffull) | (2u << 30); // hi | type=2
  v8i g1;
  g1[0] = (int)((data_size_code << 16)
              | (1u << 20)                                // pad_enable
              | (pad_interval_code << 22)
              | (pad_amount_code << 25));
  g1[1] = (int)((tensor_d0 & 0xffffu) << 16);
  g1[2] = (int)(((tensor_d0 >> 16) & 0xffffu) | ((tensor_d1 & 0xffffu) << 16));
  g1[3] = (int)(((tensor_d1 >> 16) & 0xffffu) | (tile_d0 << 16));
  g1[4] = (int)tile_d1;                                   // tile_dim2 = 0
  g1[5] = (int)(stride0_elems & 0xffffffffull);
  g1[6] = (int)((stride0_elems >> 32) & 0xffffull);       // stride1 = 0
  g1[7] = 0;
  v4i z4 = {0, 0, 0, 0};
  v8i z8 = {0, 0, 0, 0, 0, 0, 0, 0};
  __builtin_amdgcn_tensor_load_to_lds(g0, g1, z4, z4, z8, 0);
}

__device__ __forceinline__ unsigned lds_offset_of(const void* p) {
  // generic pointer to LDS: low 32 bits are the LDS offset (aperture layout)
  return (unsigned)(unsigned long long)p;
}

// ---------------------------------------------------------------------------
// Kernel 0: x_c = x - b_dec, pre-split into packed bf16 hi/lo dword arrays
// ---------------------------------------------------------------------------
__global__ __launch_bounds__(256) void sae_xc(const float* __restrict__ x,
                                              const float* __restrict__ b_dec,
                                              unsigned* __restrict__ xch,
                                              unsigned* __restrict__ xcl)
{
  const int row = blockIdx.x;
  const int tid = threadIdx.x;
#pragma unroll
  for (int i = 0; i < 2; i++) {
    int p = tid + i * 256;               // pair index 0..511
    int d = p * 2;
    float v0 = x[(size_t)row * D_IN + d]     - b_dec[d];
    float v1 = x[(size_t)row * D_IN + d + 1] - b_dec[d + 1];
    unsigned hi, lo;
    split_pair(v0, v1, hi, lo);
    xch[(size_t)row * (D_IN / 2) + p] = hi;
    xcl[(size_t)row * (D_IN / 2) + p] = lo;
  }
}

// ---------------------------------------------------------------------------
// Kernel 1: encoder GEMM  pre_acts = relu(x_c @ W_enc + b_enc)
// Block tile 128(M) x 64(N), K-step 32. 8 waves as 4(M) x 2(N); each wave
// computes a 32x32 patch = 2x2 WMMA accumulators.
// A (bf16 hi/lo): TDM, double-buffered; LDS rows 32 elems + 8-elem pad = 40.
// B: registers->split->transposed bf16 LDS [n][kpair], rows 16+4 = 20 dwords.
// ---------------------------------------------------------------------------
#define BM   128
#define BN   64
#define KT   32
#define LDAH 40   // A row stride in bf16 elems (80B, 16B-multiple)
#define LDBD 20   // B row stride in dwords (80B)

__global__ __launch_bounds__(256) void sae_encode_gemm(
    const unsigned short* __restrict__ xch,
    const unsigned short* __restrict__ xcl,
    const float* __restrict__ W_enc,
    const float* __restrict__ b_enc, float* __restrict__ pre_acts)
{
  __shared__ unsigned short AldsH[2][BM * LDAH];  // 20KB
  __shared__ unsigned short AldsL[2][BM * LDAH];  // 20KB
  __shared__ unsigned       BhiT[2][BN * LDBD];   // 10KB
  __shared__ unsigned       BloT[2][BN * LDBD];   // 10KB

  const int tid  = threadIdx.x;
  const int lane = tid & 31;
  const int wave = tid >> 5;
  const int wm   = wave & 3;                 // wave row 0..3
  const int wn   = wave >> 2;                // wave col 0..1
  const int blockM = blockIdx.x * BM;        // M fastest-varying
  const int blockN = blockIdx.y * BN;
  const int NS = D_IN / KT;                  // 32 stages

  v8f acc[2][2];
#pragma unroll
  for (int i = 0; i < 2; i++)
#pragma unroll
    for (int j = 0; j < 2; j++)
      acc[i][j] = (v8f){0.f,0.f,0.f,0.f,0.f,0.f,0.f,0.f};

  // --- A tiles via TDM (bf16, 128x32 elems, padded rows of 40 elems) -------
  auto issue_A = [&](int s) {
    const int gk0 = s * KT;
    const int buf = s & 1;
    tdm_load_2d(lds_offset_of(&AldsH[buf][0]),
                xch + (size_t)blockM * D_IN + gk0, /*2B*/ 1u,
                D_IN, BATCH, KT, BM, (unsigned long long)D_IN,
                /*pad every 16 dwords*/ 3u, /*by 4 dwords*/ 3u);
    tdm_load_2d(lds_offset_of(&AldsL[buf][0]),
                xcl + (size_t)blockM * D_IN + gk0, /*2B*/ 1u,
                D_IN, BATCH, KT, BM, (unsigned long long)D_IN,
                3u, 3u);
  };

  // --- B tile: coalesced global loads (n fastest), prefetched a stage ahead
  const int bn_n = tid & 63;                 // column within tile
  const int jb   = (tid >> 6) << 2;          // k-pair base: 0,4,8,12
  float breg[8];
  auto load_breg = [&](int s) {
    const int gk0 = s * KT;
#pragma unroll
    for (int i = 0; i < 4; i++) {
      const float* p = W_enc + (size_t)(gk0 + 2 * (jb + i)) * NLAT + blockN + bn_n;
      breg[2 * i]     = p[0];
      breg[2 * i + 1] = p[NLAT];
    }
  };
  auto convert_store_B = [&](int buf) {
#pragma unroll
    for (int i = 0; i < 4; i++) {
      unsigned hi, lo;
      split_pair(breg[2 * i], breg[2 * i + 1], hi, lo);
      BhiT[buf][bn_n * LDBD + jb + i] = hi;
      BloT[buf][bn_n * LDBD + jb + i] = lo;
    }
  };

  load_breg(0);
  if (wave == 0) issue_A(0);

  for (int s = 0; s < NS; s++) {
    const int buf = s & 1;
    convert_store_B(buf);                         // B(s) -> LDS (transposed)
    if (wave == 0) {
      if (s + 1 < NS) {
        issue_A(s + 1);                           // prefetch A into other buf
        __builtin_amdgcn_s_wait_tensorcnt(2);     // A(s)'s pair complete
      } else {
        __builtin_amdgcn_s_wait_tensorcnt(0);
      }
    }
    __syncthreads();                              // TDM A + B stores visible

    if (s + 1 < NS) load_breg(s + 1);             // overlap with WMMA below

    // ---- fragments: pure vector LDS loads ---------------------------------
    // A 16x32 bf16: lanes 0-15 -> M=lane, akb=0; lanes 16-31 -> akb=8;
    //               elems e=0..7 <-> K=akb+e, e=8..15 <-> K=16+akb+(e-8)
    V16 ahi[2], alo[2], bhi[2], blo[2];
    const int mrow = lane & 15;
    const int akb  = (lane >> 4) << 3;            // 0 or 8 (elems)
#pragma unroll
    for (int tm = 0; tm < 2; tm++) {
      const unsigned short* ph = &AldsH[buf][(wm * 32 + tm * 16 + mrow) * LDAH + akb];
      const unsigned short* pl = &AldsL[buf][(wm * 32 + tm * 16 + mrow) * LDAH + akb];
      ahi[tm].h[0] = *(const v8us*)ph;
      ahi[tm].h[1] = *(const v8us*)(ph + 16);
      alo[tm].h[0] = *(const v8us*)pl;
      alo[tm].h[1] = *(const v8us*)(pl + 16);
    }
    // B 32x16 bf16: lanes 0-15 -> N=lane, K as A (table order, kb=0); lanes 16-31 -> kb=8
    const int ncol = lane & 15;
    const int bkd  = (lane >> 4) << 2;            // dword offset: 0 or 4
#pragma unroll
    for (int tn = 0; tn < 2; tn++) {
      const unsigned* qh = &BhiT[buf][(wn * 32 + tn * 16 + ncol) * LDBD + bkd];
      const unsigned* ql = &BloT[buf][(wn * 32 + tn * 16 + ncol) * LDBD + bkd];
      bhi[tn].h[0] = *(const v8us*)qh;
      bhi[tn].h[1] = *(const v8us*)(qh + 8);
      blo[tn].h[0] = *(const v8us*)ql;
      blo[tn].h[1] = *(const v8us*)(ql + 8);
    }

    // ---- 3-pass split-bf16 WMMA: hi*hi + hi*lo + lo*hi --------------------
#pragma unroll
    for (int tm = 0; tm < 2; tm++)
#pragma unroll
      for (int tn = 0; tn < 2; tn++) {
        acc[tm][tn] = __builtin_amdgcn_wmma_f32_16x16x32_bf16(
            false, ahi[tm].b, false, bhi[tn].b, (short)0, acc[tm][tn], false, false);
        acc[tm][tn] = __builtin_amdgcn_wmma_f32_16x16x32_bf16(
            false, ahi[tm].b, false, blo[tn].b, (short)0, acc[tm][tn], false, false);
        acc[tm][tn] = __builtin_amdgcn_wmma_f32_16x16x32_bf16(
            false, alo[tm].b, false, bhi[tn].b, (short)0, acc[tm][tn], false, false);
      }
    __syncthreads();                              // release buffers
  }

  // ---- epilogue: + b_enc, relu, non-temporal store ------------------------
  // C layout: VGPR r -> lanes 0-15: M=r, N=lane; lanes 16-31: M=r+8, N=lane-16
  const int moff = (lane >> 4) << 3;
#pragma unroll
  for (int tm = 0; tm < 2; tm++)
#pragma unroll
    for (int tn = 0; tn < 2; tn++) {
      int gcol = blockN + wn * 32 + tn * 16 + (lane & 15);
      float be = b_enc[gcol];
#pragma unroll
      for (int r = 0; r < 8; r++) {
        int grow = blockM + wm * 32 + tm * 16 + moff + r;
        float v = acc[tm][tn][r] + be;
        v = v > 0.f ? v : 0.f;
        __builtin_nontemporal_store(v, &pre_acts[(size_t)grow * NLAT + gcol]);
      }
    }
}

// ---------------------------------------------------------------------------
// Kernel 2: per-row top-32 (jax.lax.top_k tie order: value desc, index asc),
// then fused sparse decode + sae_out + sum(e^2) accumulation.
// ---------------------------------------------------------------------------
__global__ __launch_bounds__(256) void sae_topk_decode(
    const float* __restrict__ pre_acts, const float* __restrict__ W_dec,
    const float* __restrict__ b_dec, const float* __restrict__ x,
    float* __restrict__ sae_out, float* __restrict__ top_acts,
    int* __restrict__ top_idx, float* __restrict__ ws)
{
  __shared__ float cval[256 * TOPK];
  __shared__ int   cidx[256 * TOPK];
  const int tid = threadIdx.x;
  const int row = blockIdx.x;

  float* mv = &cval[tid * TOPK];
  int*   mi = &cidx[tid * TOPK];
#pragma unroll
  for (int k = 0; k < TOPK; k++) { mv[k] = -1.f; mi[k] = 0x7fffffff; }

  const float* pr = pre_acts + (size_t)row * NLAT;
  for (int j = tid; j < NLAT; j += 256) {
    float v = __builtin_nontemporal_load(&pr[j]);
    if (v > mv[TOPK - 1]) {
      int p = TOPK - 1;
      while (p > 0 && v > mv[p - 1]) { mv[p] = mv[p - 1]; mi[p] = mi[p - 1]; p--; }
      mv[p] = v; mi[p] = j;
    }
  }
  __syncthreads();

  for (int s = 1; s < 256; s <<= 1) {
    if ((tid & (2 * s - 1)) == 0) {
      float ov[TOPK]; int oi[TOPK];
      const float* av = &cval[tid * TOPK];       const int* ai = &cidx[tid * TOPK];
      const float* bv = &cval[(tid + s) * TOPK]; const int* bi = &cidx[(tid + s) * TOPK];
      int ia = 0, ib = 0;
#pragma unroll
      for (int k = 0; k < TOPK; k++) {
        float va = av[ia], vb = bv[ib];
        int   xa = ai[ia], xb = bi[ib];
        bool takeA = (va > vb) || (va == vb && xa < xb);
        ov[k] = takeA ? va : vb;
        oi[k] = takeA ? xa : xb;
        ia += takeA ? 1 : 0; ib += takeA ? 0 : 1;
      }
#pragma unroll
      for (int k = 0; k < TOPK; k++) { cval[tid * TOPK + k] = ov[k]; cidx[tid * TOPK + k] = oi[k]; }
    }
    __syncthreads();
  }

  if (tid < TOPK) {
    top_acts[row * TOPK + tid] = cval[tid];
    top_idx [row * TOPK + tid] = cidx[tid];
  }

  float accd[4] = {0.f, 0.f, 0.f, 0.f};
  for (int k = 0; k < TOPK; k++) {
    float a  = cval[k];
    int   id = cidx[k];
    const float* wr = W_dec + (size_t)id * D_IN;
#pragma unroll
    for (int i = 0; i < 4; i++) accd[i] += a * wr[tid + i * 256];
  }
  float e2 = 0.f;
#pragma unroll
  for (int i = 0; i < 4; i++) {
    int d = tid + i * 256;
    float so = accd[i] + b_dec[d];
    sae_out[(size_t)row * D_IN + d] = so;
    float e = x[(size_t)row * D_IN + d] - so;
    e2 += e * e;
  }
  __syncthreads();
  cval[tid] = e2;
  __syncthreads();
  for (int s = 128; s > 0; s >>= 1) {
    if (tid < s) cval[tid] += cval[tid + s];
    __syncthreads();
  }
  if (tid == 0) atomicAdd(&ws[0], cval[0]);
}

// ---------------------------------------------------------------------------
// Kernel 3: column sums of x and sum(x^2) -> ws[2..1025], ws[1]
// ---------------------------------------------------------------------------
__global__ __launch_bounds__(256) void sae_colsum(const float* __restrict__ x,
                                                  float* __restrict__ ws)
{
  __shared__ float red[256];
  const int tid = threadIdx.x;
  const int r0  = blockIdx.x * 64;
  float cs[4] = {0.f, 0.f, 0.f, 0.f};
  float sx2 = 0.f;
  for (int r = 0; r < 64; r++) {
    const float* xr = x + (size_t)(r0 + r) * D_IN;
#pragma unroll
    for (int i = 0; i < 4; i++) {
      float v = xr[tid + i * 256];
      cs[i] += v; sx2 += v * v;
    }
  }
#pragma unroll
  for (int i = 0; i < 4; i++) atomicAdd(&ws[2 + tid + i * 256], cs[i]);
  red[tid] = sx2;
  __syncthreads();
  for (int s = 128; s > 0; s >>= 1) {
    if (tid < s) red[tid] += red[tid + s];
    __syncthreads();
  }
  if (tid == 0) atomicAdd(&ws[1], red[0]);
}

// ---------------------------------------------------------------------------
// Kernel 4: fvu = sum_e2 / (sum_x2 - sum(colsum^2)/B); auxk = multi = 0
// ---------------------------------------------------------------------------
__global__ __launch_bounds__(256) void sae_fvu(const float* __restrict__ ws,
                                               float* __restrict__ out_sc)
{
  __shared__ float red[256];
  const int tid = threadIdx.x;
  float s = 0.f;
#pragma unroll
  for (int i = 0; i < 4; i++) { float c = ws[2 + tid + i * 256]; s += c * c; }
  red[tid] = s;
  __syncthreads();
  for (int st = 128; st > 0; st >>= 1) {
    if (tid < st) red[tid] += red[tid + st];
    __syncthreads();
  }
  if (tid == 0) {
    float tv = ws[1] - red[0] / (float)BATCH;
    out_sc[0] = ws[0] / tv;   // fvu
    out_sc[1] = 0.f;          // auxk_loss
    out_sc[2] = 0.f;          // multi_topk_fvu
  }
}

__global__ void sae_ws_init(float* ws) {
  int t = blockIdx.x * 256 + threadIdx.x;
  if (t < 1026) ws[t] = 0.f;
}

// ---------------------------------------------------------------------------
// d_out flat layout (reference return order, fp32 elements):
//   [0)        sae_out 4096*1024 | [4194304) top_acts | [4325376) top_indices
//   [4456448)  fvu, auxk, multi  | [4456451) pre_acts 4096*65536
// ws: [0] sum_e2, [1] sum_x2, [2..1025] colsum,
//     [1032..)   xc_hi  (4096x1024 bf16, 8MB), then xc_lo (8MB)
// ---------------------------------------------------------------------------
extern "C" void kernel_launch(void* const* d_in, const int* in_sizes, int n_in,
                              void* d_out, int out_size, void* d_ws, size_t ws_size,
                              hipStream_t stream) {
  const float* x     = (const float*)d_in[0];
  const float* W_enc = (const float*)d_in[1];
  const float* b_enc = (const float*)d_in[2];
  const float* W_dec = (const float*)d_in[3];
  const float* b_dec = (const float*)d_in[4];

  float* outf = (float*)d_out;
  float* wsf  = (float*)d_ws;
  unsigned* xch = (unsigned*)(wsf + 1032);                 // 16B-aligned
  unsigned* xcl = xch + (size_t)BATCH * (D_IN / 2);

  const size_t TA_OFF = (size_t)BATCH * D_IN;              // 4194304
  const size_t TI_OFF = TA_OFF + (size_t)BATCH * TOPK;     // 4325376
  const size_t SC_OFF = TI_OFF + (size_t)BATCH * TOPK;     // 4456448
  const size_t PA_OFF = SC_OFF + 3;                        // 4456451

  sae_ws_init<<<5, 256, 0, stream>>>(wsf);
  sae_xc<<<BATCH, 256, 0, stream>>>(x, b_dec, xch, xcl);

  dim3 ggrid(BATCH / BM, NLAT / BN);                       // M fastest: 32 x 1024
  sae_encode_gemm<<<ggrid, 256, 0, stream>>>(
      (const unsigned short*)xch, (const unsigned short*)xcl,
      W_enc, b_enc, outf + PA_OFF);

  sae_colsum<<<64, 256, 0, stream>>>(x, wsf);

  sae_topk_decode<<<BATCH, 256, 0, stream>>>(outf + PA_OFF, W_dec, b_dec, x,
                                             outf, outf + TA_OFF,
                                             (int*)(outf + TI_OFF), wsf);

  sae_fvu<<<1, 256, 0, stream>>>(wsf, outf + SC_OFF);
}
